// EEVTemporalModel_4045859193682
// MI455X (gfx1250) — matrix operand order repaired and checked by reading, not compiled, not run on hardware
//
#include <hip/hip_runtime.h>

// ---------------------------------------------------------------------------
// Fused 2-layer LSTM (B=1024, T=2048, D=15, H=64) + FC(4) + softmax for
// MI455X (gfx1250, wave32, WMMA).
//
// Latency-bound serial recurrence -> persistent workgroup per 32 batch rows,
// whole sequence in one kernel. Per step: gates = [h | x] @ [W_hh ; W_ih]^T
// via v_wmma_f32_16x16x32_f16 with register-resident f16 weight fragments and
// inline-0 C seed; biases are folded into the activation exp arguments (FMA).
// i/f/g/o tiles for the same element land in the same lane, so the cell
// update is per-lane VALU using raw v_exp_f32 / v_rcp_f32 that co-execute
// with the WMMA pipe. Double-buffered LDS staging -> 2 barriers per timestep.
// ---------------------------------------------------------------------------

typedef _Float16 v16h __attribute__((ext_vector_type(16)));
typedef _Float16 v8h  __attribute__((ext_vector_type(8)));
typedef float    v8f  __attribute__((ext_vector_type(8)));

#define WMMA_F16(a, b, c) \
  __builtin_amdgcn_wmma_f32_16x16x32_f16(false, (a), false, (b), (short)0, (c), false, false)

namespace {
constexpr int D    = 15;    // INPUT_DIM
constexpr int H    = 64;    // HIDDEN
constexpr int T    = 2048;
constexpr int Btot = 1024;
constexpr int NC   = 4;
constexpr int TILE = 32;    // batch rows per workgroup
constexpr int P0   = 96;    // LDS pitch (halves) for [h0 | x | pad], K=96
constexpr int P1   = 128;   // LDS pitch (halves) for [h1 | h0],     K=128
constexpr float LOG2E = 1.4426950408889634f;
}

// sigmoid(x + b) = rcp(1 + exp2(fma(x, -log2e, off))), off = -b*log2e
__device__ inline float fsig_b(float x, float off) {
  float e = __builtin_amdgcn_exp2f(__builtin_fmaf(x, -LOG2E, off));
  return __builtin_amdgcn_rcpf(1.0f + e);
}
// tanh(x + b) = 1 - 2*rcp(exp2(fma(x, 2log2e, off)) + 1), off = 2*b*log2e
__device__ inline float ftanh_b(float x, float off) {
  float e = __builtin_amdgcn_exp2f(__builtin_fmaf(x, 2.0f * LOG2E, off));
  float r = __builtin_amdgcn_rcpf(e + 1.0f);
  return __builtin_fmaf(-2.0f, r, 1.0f);
}
// tanh(x) for the cell output path (no bias)
__device__ inline float ftanh(float x) {
  float e = __builtin_amdgcn_exp2f(x * (2.0f * LOG2E));
  float r = __builtin_amdgcn_rcpf(e + 1.0f);
  return __builtin_fmaf(-2.0f, r, 1.0f);
}

// A fragment (16x32 f16): lane L -> row M = mbase + (L&15),
// halves j=0..15 -> K = kb + ((L>>4)<<4) + j.
__device__ inline v16h load_afrag(const _Float16* buf, int pitch, int mbase, int kb, int lane) {
  const int row  = mbase + (lane & 15);
  const int koff = kb + ((lane >> 4) << 4);
  v8h lo = *(const v8h*)(buf + row * pitch + koff);
  v8h hi = *(const v8h*)(buf + row * pitch + koff + 8);
  return __builtin_shufflevector(lo, hi, 0,1,2,3,4,5,6,7,8,9,10,11,12,13,14,15);
}

// B fragment (32x16 K-major): lane L -> col N = n, halves j -> K = kb + j,
// where kb already includes the (lane>>4)*16 K-half select.
// Layer0 K-space: [ w_hh (64) | w_ih (15) | zero-pad (17) ]
__device__ inline v16h load_bfrag0(const float* w_hh, const float* w_ih, int n, int kb) {
  v16h b;
#pragma unroll
  for (int j = 0; j < 16; ++j) {
    int k = kb + j;
    float v = 0.0f;
    if (k < H)              v = w_hh[n * H + k];
    else if ((k - H) < D)   v = w_ih[n * D + (k - H)];
    b[j] = (_Float16)v;
  }
  return b;
}

// Layer1 K-space: [ w_hh (64) | w_ih (64) ]
__device__ inline v16h load_bfrag1(const float* w_hh, const float* w_ih, int n, int kb) {
  v16h b;
#pragma unroll
  for (int j = 0; j < 16; ++j) {
    int k = kb + j;
    float v = (k < H) ? w_hh[n * H + k] : w_ih[n * H + (k - H)];
    b[j] = (_Float16)v;
  }
  return b;
}

__global__ __launch_bounds__(256)
void lstm2_fused_kernel(const float* __restrict__ x,
                        const float* __restrict__ w_ih0, const float* __restrict__ w_hh0,
                        const float* __restrict__ b_ih0, const float* __restrict__ b_hh0,
                        const float* __restrict__ w_ih1, const float* __restrict__ w_hh1,
                        const float* __restrict__ b_ih1, const float* __restrict__ b_hh1,
                        const float* __restrict__ fc_w,  const float* __restrict__ fc_b,
                        float* __restrict__ out) {
  // Double-buffered staging: step t reads [p = t&1], writes state for t+1
  // into [p^1] -> no write-after-read hazards, only 2 barriers per step.
  __shared__ __align__(16) _Float16 sHX[2][TILE * P0]; // [32 x (h0|x|pad)] 12 KB
  __shared__ __align__(16) _Float16 sHH[2][TILE * P1]; // [32 x (h1|h0)]   16 KB

  const int tid  = threadIdx.x;
  const int lane = tid & 31;
  const int wave = tid >> 5;
  const int ht   = wave & 3;   // hidden-column tile (16 cols) this wave owns
  const int mt   = wave >> 2;  // batch M-tile (16 rows) this wave owns
  const int rowBase = blockIdx.x * TILE;

  // ---- Register-resident f16 weight fragments + folded biases (once) ----
  v16h b0[4][3];   // [gate][kstep], K = 96
  v16h b1[4][4];   // [gate][kstep], K = 128
  float off0[4], off1[4]; // activation-folded bias constants per gate
  const int khalf = ((lane >> 4) << 4); // 0 or 16: K-half owned by this lane
#pragma unroll
  for (int g = 0; g < 4; ++g) {
    const int n = g * H + ht * 16 + (lane & 15); // gate-matrix row / output col
#pragma unroll
    for (int ks = 0; ks < 3; ++ks) b0[g][ks] = load_bfrag0(w_hh0, w_ih0, n, ks * 32 + khalf);
#pragma unroll
    for (int ks = 0; ks < 4; ++ks) b1[g][ks] = load_bfrag1(w_hh1, w_ih1, n, ks * 32 + khalf);
    const float bb0 = b_ih0[n] + b_hh0[n];
    const float bb1 = b_ih1[n] + b_hh1[n];
    // gates i,f,o (sigmoid): off = -b*log2e ; gate g (tanh): off = 2*b*log2e
    off0[g] = (g == 2) ? (2.0f * LOG2E * bb0) : (-LOG2E * bb0);
    off1[g] = (g == 2) ? (2.0f * LOG2E * bb1) : (-LOG2E * bb1);
  }

  // ---- Zero LDS state (h0 = h1 = 0, pad cols = 0), stage x_0 ----
  for (int i = tid; i < 2 * TILE * P0; i += 256) (&sHX[0][0])[i] = (_Float16)0.0f;
  for (int i = tid; i < 2 * TILE * P1; i += 256) (&sHH[0][0])[i] = (_Float16)0.0f;
  for (int i = tid; i < TILE * D; i += 256) {
    const int r = i / D, d = i - r * D;
    sHX[0][r * P0 + H + d] = (_Float16)x[((size_t)(rowBase + r) * T + 0) * D + d];
  }

  v8f c0 = {}; // cell state, layer 0 (lane-distributed in C layout)
  v8f c1 = {}; // cell state, layer 1
  __syncthreads();

  const int col   = ht * 16 + (lane & 15);          // hidden column (C/D layout)
  const int mbase = mt * 16 + ((lane >> 4) << 3);   // M row base (C/D layout)

  for (int t = 0; t < T; ++t) {
    const int p = t & 1, q = p ^ 1;
    const _Float16* rHX = sHX[p];
    _Float16*       wHX = sHX[q];
    const _Float16* rHH = sHH[p];
    _Float16*       wHH = sHH[q];

    // ================= Layer 0: gates = [h0 | x_t] @ W0cat^T ==============
    v16h a0[3];
#pragma unroll
    for (int ks = 0; ks < 3; ++ks) a0[ks] = load_afrag(rHX, P0, mt * 16, ks * 32, lane);

    // Stage x_{t+1} into the back buffer (overlaps with WMMA latency).
    if (t + 1 < T) {
      for (int i = tid; i < TILE * D; i += 256) {
        const int r = i / D, d = i - r * D;
        wHX[r * P0 + H + d] =
            (_Float16)x[((size_t)(rowBase + r) * T + (t + 1)) * D + d];
      }
      if (t + 2 < T && tid < TILE) // pull t+2 rows toward the WGP
        __builtin_prefetch(&x[((size_t)(rowBase + tid) * T + (t + 2)) * D], 0, 1);
    }

    v8f acc0[4];
#pragma unroll
    for (int g = 0; g < 4; ++g) {
      v8f a = WMMA_F16(a0[0], b0[g][0], (v8f){}); // C = inline 0
#pragma unroll
      for (int ks = 1; ks < 3; ++ks) a = WMMA_F16(a0[ks], b0[g][ks], a);
      acc0[g] = a;
    }

    v8f h0v;
#pragma unroll
    for (int e = 0; e < 8; ++e) {
      const float iv = fsig_b(acc0[0][e], off0[0]);
      const float fv = fsig_b(acc0[1][e], off0[1]);
      const float gv = ftanh_b(acc0[2][e], off0[2]);
      const float ov = fsig_b(acc0[3][e], off0[3]);
      const float c  = __builtin_fmaf(fv, c0[e], iv * gv);
      c0[e]  = c;
      h0v[e] = ov * ftanh(c);
    }
    // h0_t -> back buffer (layer0 A of t+1) and front sHH cols 64..127
    // (layer1 A of this step; made visible by the mid barrier).
#pragma unroll
    for (int e = 0; e < 8; ++e) {
      const _Float16 hv = (_Float16)h0v[e];
      wHX[(mbase + e) * P0 + col] = hv;
      ((_Float16*)rHH)[(mbase + e) * P1 + H + col] = hv;
    }
    __syncthreads(); // mid barrier: publish h0_t for layer 1

    // ================= Layer 1: gates = [h1 | h0_t] @ W1cat^T =============
    v16h a1[4];
#pragma unroll
    for (int ks = 0; ks < 4; ++ks) a1[ks] = load_afrag(rHH, P1, mt * 16, ks * 32, lane);

    v8f acc1[4];
#pragma unroll
    for (int g = 0; g < 4; ++g) {
      v8f a = WMMA_F16(a1[0], b1[g][0], (v8f){}); // C = inline 0
#pragma unroll
      for (int ks = 1; ks < 4; ++ks) a = WMMA_F16(a1[ks], b1[g][ks], a);
      acc1[g] = a;
    }

    v8f h1v;
#pragma unroll
    for (int e = 0; e < 8; ++e) {
      const float iv = fsig_b(acc1[0][e], off1[0]);
      const float fv = fsig_b(acc1[1][e], off1[1]);
      const float gv = ftanh_b(acc1[2][e], off1[2]);
      const float ov = fsig_b(acc1[3][e], off1[3]);
      const float c  = __builtin_fmaf(fv, c1[e], iv * gv);
      c1[e]  = c;
      h1v[e] = ov * ftanh(c);
    }
#pragma unroll
    for (int e = 0; e < 8; ++e)
      wHH[(mbase + e) * P1 + col] = (_Float16)h1v[e]; // h1_t for step t+1
    __syncthreads(); // end barrier: publish back buffers for step t+1
  }

  // ---- FC + softmax on h2[:, T-1, :] ----
  // Last step (t = T-1) scattered h1 into sHH[((T-1)&1)^1][:, 0..63].
  const _Float16* hfin = sHH[((T - 1) & 1) ^ 1];
  if (tid < TILE) {
    float logits[NC];
    float m = -3.0e38f;
#pragma unroll
    for (int c = 0; c < NC; ++c) {
      float s = fc_b[c];
#pragma unroll
      for (int k = 0; k < H; ++k)
        s += (float)hfin[tid * P1 + k] * fc_w[c * H + k];
      logits[c] = s;
      m = fmaxf(m, s);
    }
    float denom = 0.0f;
#pragma unroll
    for (int c = 0; c < NC; ++c) {
      logits[c] = __builtin_amdgcn_exp2f((logits[c] - m) * LOG2E);
      denom += logits[c];
    }
    const float inv = __builtin_amdgcn_rcpf(denom);
#pragma unroll
    for (int c = 0; c < NC; ++c)
      out[(size_t)(rowBase + tid) * NC + c] = logits[c] * inv;
  }
}

extern "C" void kernel_launch(void* const* d_in, const int* in_sizes, int n_in,
                              void* d_out, int out_size, void* d_ws, size_t ws_size,
                              hipStream_t stream) {
  (void)in_sizes; (void)n_in; (void)out_size; (void)d_ws; (void)ws_size;
  const float* x     = (const float*)d_in[0];
  const float* w_ih0 = (const float*)d_in[1];
  const float* w_hh0 = (const float*)d_in[2];
  const float* b_ih0 = (const float*)d_in[3];
  const float* b_hh0 = (const float*)d_in[4];
  const float* w_ih1 = (const float*)d_in[5];
  const float* w_hh1 = (const float*)d_in[6];
  const float* b_ih1 = (const float*)d_in[7];
  const float* b_hh1 = (const float*)d_in[8];
  const float* fc_w  = (const float*)d_in[9];
  const float* fc_b  = (const float*)d_in[10];
  float* out = (float*)d_out;

  dim3 grid(Btot / TILE); // 32 workgroups, one per 32 batch rows
  dim3 block(256);        // 8 wave32 waves
  lstm2_fused_kernel<<<grid, block, 0, stream>>>(x, w_ih0, w_hh0, b_ih0, b_hh0,
                                                 w_ih1, w_hh1, b_ih1, b_hh1,
                                                 fc_w, fc_b, out);
}
